// Net_73624329388059
// MI455X (gfx1250) — compile-verified, matrix operand order checked by reading, not run on hardware
//
#include <hip/hip_runtime.h>
#include <hip/hip_bf16.h>
#include <math.h>

typedef __attribute__((ext_vector_type(16))) _Float16 v16h;
typedef __attribute__((ext_vector_type(8)))  float    v8f;

#define DEVINL __device__ __forceinline__

// ---------------------------------------------------------------------------
// WMMA wrapper: D = A(16x32 f16) * B(32x16 f16) + C(16x16 f32)
// ---------------------------------------------------------------------------
DEVINL v8f wmma_f16(v16h a, v16h b, v8f c) {
  return __builtin_amdgcn_wmma_f32_16x16x32_f16(
      /*neg_a=*/false, a, /*neg_b=*/false, b,
      /*c_mod=*/(short)0, c, /*reuse_a=*/false, /*reuse_b=*/false);
}

// k-pattern for half-slot pair d (0..7) of a 16-bit A/B operand (wave32):
//   k0 = (d>>2)*16 + hi*8 + (d&3)*2 ; halves are k0, k0+1
// A: lane (lane&15) holds row m, halves stripe K.
// B: lane (lane&15) holds col n, halves stripe K (symmetric layout).
DEVINL int kpat(int d, int hi) {
  return ((d >> 2) << 4) + (hi << 3) + ((d & 3) << 1);
}

// Load A-tile (element (m, kbase+kpat)) from a row-major f16 matrix.
DEVINL v16h load_tile_rowmajor(const _Float16* base, int ld, int m, int kbase, int hi) {
  v16h r;
  const _Float16* rp = base + (size_t)m * ld + kbase;
#pragma unroll
  for (int d = 0; d < 8; ++d) {
    int k0 = kpat(d, hi);
    r[2 * d]     = rp[k0];
    r[2 * d + 1] = rp[k0 + 1];
  }
  return r;
}

// Load B-tile from row-major (Ktot x N) f16 weight matrix, zero-padded past Ktot.
DEVINL v16h load_btile_w(const _Float16* w, int N, int Ktot, int kbase, int nbase,
                         int lo, int hi) {
  v16h r;
#pragma unroll
  for (int d = 0; d < 8; ++d) {
    int k0 = kpat(d, hi) + kbase;
    r[2 * d]     = (k0     < Ktot) ? w[(size_t)(k0)     * N + nbase + lo] : (_Float16)0.f;
    r[2 * d + 1] = (k0 + 1 < Ktot) ? w[(size_t)(k0 + 1) * N + nbase + lo] : (_Float16)0.f;
  }
  return r;
}

DEVINL unsigned long long wave_min_u64(unsigned long long v) {
#pragma unroll
  for (int off = 16; off; off >>= 1) {
    unsigned lo = (unsigned)v, hi = (unsigned)(v >> 32);
    unsigned olo = __shfl_xor(lo, off, 32);
    unsigned ohi = __shfl_xor(hi, off, 32);
    unsigned long long o = ((unsigned long long)ohi << 32) | olo;
    if (o < v) v = o;
  }
  return v;
}

// ---------------------------------------------------------------------------
// f32 -> f16 conversion
// ---------------------------------------------------------------------------
__global__ void k_f32_to_f16(const float* __restrict__ in, _Float16* __restrict__ out, int n) {
  int i = blockIdx.x * 256 + threadIdx.x;
  if (i < n) out[i] = (_Float16)in[i];
}

// ---------------------------------------------------------------------------
// kNN for 1-d features (first EdgeConv). One wave per point.
// ---------------------------------------------------------------------------
__global__ void __launch_bounds__(256) k_knn1(const float* __restrict__ x,
                                              int* __restrict__ idxout) {
  __shared__ float sx[1024];
  int b = blockIdx.x >> 7;        // 128 blocks per batch
  int pblk = blockIdx.x & 127;
  int tid = threadIdx.x;
  const float* xb = x + (size_t)b * 1024;
  for (int i = tid; i < 1024; i += 256) sx[i] = xb[i];
  __syncthreads();

  int wave = tid >> 5, lane = tid & 31;
  int p = pblk * 8 + wave;
  float xi = sx[p];
  float d[32];
#pragma unroll
  for (int t = 0; t < 32; ++t) {
    float dx = xi - sx[t * 32 + lane];
    d[t] = dx * dx;
  }
  int* op = idxout + ((size_t)b * 1024 + p) * 16;
  for (int it = 0; it < 16; ++it) {
    unsigned long long best = ~0ull; int bt = 0;
#pragma unroll
    for (int t = 0; t < 32; ++t) {
      unsigned long long key =
          ((unsigned long long)__float_as_uint(d[t]) << 32) | (unsigned)(t * 32 + lane);
      if (key < best) { best = key; bt = t; }
    }
    unsigned long long wmin = wave_min_u64(best);
    if (best == wmin) d[bt] = __builtin_inff();
    if (lane == 0) op[it] = (int)(wmin & 0xffffffffu);
  }
}

// ---------------------------------------------------------------------------
// Squared norms of 64-dim f16 features
// ---------------------------------------------------------------------------
__global__ void k_sqnorm(const _Float16* __restrict__ xh, float* __restrict__ sq, int total) {
  int i = blockIdx.x * 256 + threadIdx.x;
  if (i < total) {
    const _Float16* r = xh + (size_t)i * 64;
    float s = 0.f;
#pragma unroll
    for (int c = 0; c < 64; ++c) { float v = (float)r[c]; s += v * v; }
    sq[i] = s;
  }
}

// ---------------------------------------------------------------------------
// kNN for 64-dim features via WMMA Gram tiles.
// One workgroup (4 waves) = one (batch, 16-row block). Distance strip in LDS.
// ---------------------------------------------------------------------------
#define DSTRIDE 1028  // 1028 % 64 == 4, 8*1028 % 64 == 32 -> conflict-free

__global__ void __launch_bounds__(128) k_knn64(const _Float16* __restrict__ xin,
                                               const float* __restrict__ sq,
                                               int* __restrict__ idxout) {
  extern __shared__ float sdist[];  // 16 * DSTRIDE floats
  int b = blockIdx.x >> 6;
  int rb = blockIdx.x & 63;
  int row0 = rb * 16;
  int tid = threadIdx.x, wave = tid >> 5, lane = tid & 31;
  int lo = lane & 15, hi = lane >> 4;

  const _Float16* xb = xin + (size_t)b * 1024 * 64;
  const float* sqb = sq + (size_t)b * 1024;

  v16h a0 = load_tile_rowmajor(xb + (size_t)row0 * 64, 64, lo, 0, hi);
  v16h a1 = load_tile_rowmajor(xb + (size_t)row0 * 64, 64, lo, 32, hi);
  float si[8];
#pragma unroll
  for (int r = 0; r < 8; ++r) si[r] = sqb[row0 + r + hi * 8];

  for (int t = 0; t < 16; ++t) {
    int col0 = (wave + t * 4) * 16;
    v16h b0 = load_tile_rowmajor(xb + (size_t)col0 * 64, 64, lo, 0, hi);
    v16h b1 = load_tile_rowmajor(xb + (size_t)col0 * 64, 64, lo, 32, hi);
    v8f acc = {};
    acc = wmma_f16(a0, b0, acc);
    acc = wmma_f16(a1, b1, acc);
    float sj = sqb[col0 + lo];
#pragma unroll
    for (int r = 0; r < 8; ++r) {
      int m = r + hi * 8;
      float dd = si[r] + sj - 2.0f * acc[r];
      sdist[m * DSTRIDE + col0 + lo] = dd < 0.f ? 0.f : dd;
    }
  }
  __syncthreads();

  for (int rw = wave; rw < 16; rw += 4) {
    float d[32];
#pragma unroll
    for (int t = 0; t < 32; ++t) d[t] = sdist[rw * DSTRIDE + t * 32 + lane];
    int* op = idxout + ((size_t)b * 1024 + row0 + rw) * 16;
    for (int it = 0; it < 16; ++it) {
      unsigned long long best = ~0ull; int bt = 0;
#pragma unroll
      for (int t = 0; t < 32; ++t) {
        unsigned long long key =
            ((unsigned long long)__float_as_uint(d[t]) << 32) | (unsigned)(t * 32 + lane);
        if (key < best) { best = key; bt = t; }
      }
      unsigned long long wmin = wave_min_u64(best);
      if (best == wmin) d[bt] = __builtin_inff();
      if (lane == 0) op[it] = (int)(wmin & 0xffffffffu);
    }
  }
}

// ---------------------------------------------------------------------------
// EdgeConv: one wave per point; 16 neighbors = the 16-row WMMA M-tile.
//   h = relu([xi, xj-xi] @ w1 + b1); out = colmax(h @ w2 + b2)
// ---------------------------------------------------------------------------
template <int CIN>
__global__ void __launch_bounds__(128) k_edgeconv(const _Float16* __restrict__ xin,
                                                  const int* __restrict__ idx,
                                                  const _Float16* __restrict__ w1h,
                                                  const float* __restrict__ b1,
                                                  const _Float16* __restrict__ w2h,
                                                  const float* __restrict__ b2,
                                                  _Float16* __restrict__ xout) {
  __shared__ _Float16 sh[4][16 * 64];
  int tid = threadIdx.x, wave = tid >> 5, lane = tid & 31;
  int lo = lane & 15, hi = lane >> 4;
  int g = blockIdx.x * 4 + wave;  // global point
  int b = g >> 10;
  int nb = idx[(size_t)g * 16 + lo];  // neighbor id for row m = lo
  const _Float16* xbase = xin + (size_t)b * 1024 * CIN;

  v8f acc[4] = {};
  if (CIN == 64) {
    const _Float16* xi = xbase + (size_t)(g & 1023) * 64;
    const _Float16* xj = xbase + (size_t)nb * 64;
#pragma unroll
    for (int c = 0; c < 4; ++c) {
      v16h a;
      if (c < 2) {
        a = load_tile_rowmajor(xi, 0, 0, c * 32, hi);  // broadcast xi over rows
      } else {
        int kb = (c - 2) * 32;
#pragma unroll
        for (int d = 0; d < 8; ++d) {
          int k0 = kpat(d, hi) + kb;
          a[2 * d]     = (_Float16)((float)xj[k0]     - (float)xi[k0]);
          a[2 * d + 1] = (_Float16)((float)xj[k0 + 1] - (float)xi[k0 + 1]);
        }
      }
#pragma unroll
      for (int nt = 0; nt < 4; ++nt) {
        v16h bt = load_btile_w(w1h, 64, 2 * CIN, c * 32, nt * 16, lo, hi);
        acc[nt] = wmma_f16(a, bt, acc[nt]);
      }
    }
  } else {  // CIN == 1 : edge feature is [xi, xj-xi], K=2 (zero-padded to 32)
    float xi = (float)xbase[g & 1023];
    float xj = (float)xbase[nb];
    v16h a = {};
    if (hi == 0) { a[0] = (_Float16)xi; a[1] = (_Float16)(xj - xi); }
#pragma unroll
    for (int nt = 0; nt < 4; ++nt) {
      v16h bt = load_btile_w(w1h, 64, 2, 0, nt * 16, lo, hi);
      acc[nt] = wmma_f16(a, bt, acc[nt]);
    }
  }

  // bias + relu -> LDS (16x64 f16, row-major), wave-private region
  _Float16* hb = sh[wave];
#pragma unroll
  for (int nt = 0; nt < 4; ++nt) {
    float bb = b1[nt * 16 + lo];
#pragma unroll
    for (int r = 0; r < 8; ++r) {
      float v = acc[nt][r] + bb;
      v = v > 0.f ? v : 0.f;
      hb[(r + hi * 8) * 64 + nt * 16 + lo] = (_Float16)v;
    }
  }
  asm volatile("s_wait_dscnt 0" ::: "memory");

  // layer 2: K=64, N=64
  v8f acc2[4] = {};
#pragma unroll
  for (int c = 0; c < 2; ++c) {
    v16h a2 = load_tile_rowmajor(hb, 64, lo, c * 32, hi);
#pragma unroll
    for (int nt = 0; nt < 4; ++nt) {
      v16h bt = load_btile_w(w2h, 64, 64, c * 32, nt * 16, lo, hi);
      acc2[nt] = wmma_f16(a2, bt, acc2[nt]);
    }
  }

  // bias + max over the 16 neighbor rows
#pragma unroll
  for (int nt = 0; nt < 4; ++nt) {
    float bb = b2[nt * 16 + lo];
    float vm = -3.4e38f;
#pragma unroll
    for (int r = 0; r < 8; ++r) {
      float v = acc2[nt][r] + bb;
      vm = v > vm ? v : vm;
    }
    float o = __shfl_xor(vm, 16, 32);
    vm = o > vm ? o : vm;
    if (hi == 0) xout[(size_t)g * 64 + nt * 16 + lo] = (_Float16)vm;
  }
}

// ---------------------------------------------------------------------------
// Final MLP: 192 -> 128 -> 64 -> 32 -> 2 + log_softmax. One wave per 16 points.
// Concat(x1,x2,x3) is read directly from the three f16 buffers (no concat copy).
// ---------------------------------------------------------------------------
__global__ void __launch_bounds__(128) k_mlp(const _Float16* __restrict__ x1h,
                                             const _Float16* __restrict__ x2h,
                                             const _Float16* __restrict__ x3h,
                                             const _Float16* __restrict__ mw1h,
                                             const float* __restrict__ mb1,
                                             const _Float16* __restrict__ mw2h,
                                             const float* __restrict__ mb2,
                                             const _Float16* __restrict__ mw3h,
                                             const float* __restrict__ mb3,
                                             const float* __restrict__ mw4,
                                             const float* __restrict__ mb4,
                                             float* __restrict__ out) {
  __shared__ _Float16 sh[4][16 * (128 + 64 + 32)];
  int tid = threadIdx.x, wave = tid >> 5, lane = tid & 31;
  int lo = lane & 15, hi = lane >> 4;
  int p0 = (blockIdx.x * 4 + wave) * 16;

  _Float16* h1 = sh[wave];
  _Float16* h2 = h1 + 16 * 128;
  _Float16* h3 = h2 + 16 * 64;

  const _Float16* srcs[3] = {x1h, x2h, x3h};

  // layer 1: K=192 (6 chunks), N=128 (8 tiles)
  v8f acc1[8] = {};
#pragma unroll
  for (int c = 0; c < 6; ++c) {
    const _Float16* src = srcs[c >> 1];
    v16h a = load_tile_rowmajor(src + (size_t)p0 * 64, 64, lo, (c & 1) * 32, hi);
#pragma unroll
    for (int nt = 0; nt < 8; ++nt) {
      v16h bt = load_btile_w(mw1h, 128, 192, c * 32, nt * 16, lo, hi);
      acc1[nt] = wmma_f16(a, bt, acc1[nt]);
    }
  }
#pragma unroll
  for (int nt = 0; nt < 8; ++nt) {
    float bb = mb1[nt * 16 + lo];
#pragma unroll
    for (int r = 0; r < 8; ++r) {
      float v = acc1[nt][r] + bb;
      v = v > 0.f ? v : 0.f;
      h1[(r + hi * 8) * 128 + nt * 16 + lo] = (_Float16)v;
    }
  }
  asm volatile("s_wait_dscnt 0" ::: "memory");

  // layer 2: K=128, N=64
  v8f acc2[4] = {};
#pragma unroll
  for (int c = 0; c < 4; ++c) {
    v16h a = load_tile_rowmajor(h1, 128, lo, c * 32, hi);
#pragma unroll
    for (int nt = 0; nt < 4; ++nt) {
      v16h bt = load_btile_w(mw2h, 64, 128, c * 32, nt * 16, lo, hi);
      acc2[nt] = wmma_f16(a, bt, acc2[nt]);
    }
  }
#pragma unroll
  for (int nt = 0; nt < 4; ++nt) {
    float bb = mb2[nt * 16 + lo];
#pragma unroll
    for (int r = 0; r < 8; ++r) {
      float v = acc2[nt][r] + bb;
      v = v > 0.f ? v : 0.f;
      h2[(r + hi * 8) * 64 + nt * 16 + lo] = (_Float16)v;
    }
  }
  asm volatile("s_wait_dscnt 0" ::: "memory");

  // layer 3: K=64, N=32
  v8f acc3[2] = {};
#pragma unroll
  for (int c = 0; c < 2; ++c) {
    v16h a = load_tile_rowmajor(h2, 64, lo, c * 32, hi);
#pragma unroll
    for (int nt = 0; nt < 2; ++nt) {
      v16h bt = load_btile_w(mw3h, 32, 64, c * 32, nt * 16, lo, hi);
      acc3[nt] = wmma_f16(a, bt, acc3[nt]);
    }
  }
#pragma unroll
  for (int nt = 0; nt < 2; ++nt) {
    float bb = mb3[nt * 16 + lo];
#pragma unroll
    for (int r = 0; r < 8; ++r) {
      float v = acc3[nt][r] + bb;
      v = v > 0.f ? v : 0.f;
      h3[(r + hi * 8) * 32 + nt * 16 + lo] = (_Float16)v;
    }
  }
  asm volatile("s_wait_dscnt 0" ::: "memory");

  // layer 4 (32 -> 2) + log_softmax, lanes 0..15 each own one point
  if (hi == 0) {
    float o0 = mb4[0], o1 = mb4[1];
#pragma unroll
    for (int k = 0; k < 32; ++k) {
      float v = (float)h3[lo * 32 + k];
      o0 += v * mw4[k * 2 + 0];
      o1 += v * mw4[k * 2 + 1];
    }
    float mx = o0 > o1 ? o0 : o1;
    float ls = mx + logf(expf(o0 - mx) + expf(o1 - mx));
    out[(size_t)(p0 + lo) * 2 + 0] = o0 - ls;
    out[(size_t)(p0 + lo) * 2 + 1] = o1 - ls;
  }
}

// ---------------------------------------------------------------------------
// Host-side orchestration
// ---------------------------------------------------------------------------
extern "C" void kernel_launch(void* const* d_in, const int* in_sizes, int n_in,
                              void* d_out, int out_size, void* d_ws, size_t ws_size,
                              hipStream_t stream) {
  const float* x    = (const float*)d_in[0];
  const float* c1w1 = (const float*)d_in[1];
  const float* c1b1 = (const float*)d_in[2];
  const float* c1w2 = (const float*)d_in[3];
  const float* c1b2 = (const float*)d_in[4];
  const float* c2w1 = (const float*)d_in[5];
  const float* c2b1 = (const float*)d_in[6];
  const float* c2w2 = (const float*)d_in[7];
  const float* c2b2 = (const float*)d_in[8];
  const float* c3w1 = (const float*)d_in[9];
  const float* c3b1 = (const float*)d_in[10];
  const float* c3w2 = (const float*)d_in[11];
  const float* c3b2 = (const float*)d_in[12];
  const float* mw1  = (const float*)d_in[13];
  const float* mb1  = (const float*)d_in[14];
  const float* mw2  = (const float*)d_in[15];
  const float* mb2  = (const float*)d_in[16];
  const float* mw3  = (const float*)d_in[17];
  const float* mb3  = (const float*)d_in[18];
  const float* mw4  = (const float*)d_in[19];
  const float* mb4  = (const float*)d_in[20];
  float* out = (float*)d_out;

  char* ws = (char*)d_ws;
  size_t off = 0;
  auto alloc = [&](size_t bytes) -> void* {
    void* p = ws + off;
    off = (off + bytes + 255) & ~(size_t)255;
    return p;
  };

  const int NPT = 64 * 1024;  // total points
  _Float16* xh    = (_Float16*)alloc((size_t)NPT * 2);
  _Float16* x1h   = (_Float16*)alloc((size_t)NPT * 64 * 2);
  _Float16* x2h   = (_Float16*)alloc((size_t)NPT * 64 * 2);
  _Float16* x3h   = (_Float16*)alloc((size_t)NPT * 64 * 2);
  float*    sqbuf = (float*)alloc((size_t)NPT * 4);
  int*      idxb  = (int*)alloc((size_t)NPT * 16 * 4);
  _Float16* c1w1h = (_Float16*)alloc(128 * 2);
  _Float16* c1w2h = (_Float16*)alloc(4096 * 2);
  _Float16* c2w1h = (_Float16*)alloc(8192 * 2);
  _Float16* c2w2h = (_Float16*)alloc(4096 * 2);
  _Float16* c3w1h = (_Float16*)alloc(8192 * 2);
  _Float16* c3w2h = (_Float16*)alloc(4096 * 2);
  _Float16* mw1h  = (_Float16*)alloc(24576 * 2);
  _Float16* mw2h  = (_Float16*)alloc(8192 * 2);
  _Float16* mw3h  = (_Float16*)alloc(2048 * 2);

  auto cvt = [&](const float* src, _Float16* dst, int n) {
    k_f32_to_f16<<<(n + 255) / 256, 256, 0, stream>>>(src, dst, n);
  };
  cvt(x, xh, NPT);
  cvt(c1w1, c1w1h, 128);
  cvt(c1w2, c1w2h, 4096);
  cvt(c2w1, c2w1h, 8192);
  cvt(c2w2, c2w2h, 4096);
  cvt(c3w1, c3w1h, 8192);
  cvt(c3w2, c3w2h, 4096);
  cvt(mw1, mw1h, 24576);
  cvt(mw2, mw2h, 8192);
  cvt(mw3, mw3h, 2048);

  // ---- EdgeConv 1 (1-d input) ----
  k_knn1<<<64 * 128, 256, 0, stream>>>(x, idxb);
  k_edgeconv<1><<<NPT / 4, 128, 0, stream>>>(xh, idxb, c1w1h, c1b1, c1w2h, c1b2, x1h);

  // ---- EdgeConv 2 ----
  const int knn64_lds = 16 * DSTRIDE * 4;
  hipFuncSetAttribute((const void*)k_knn64,
                      hipFuncAttributeMaxDynamicSharedMemorySize, knn64_lds);
  k_sqnorm<<<NPT / 256, 256, 0, stream>>>(x1h, sqbuf, NPT);
  k_knn64<<<64 * 64, 128, knn64_lds, stream>>>(x1h, sqbuf, idxb);
  k_edgeconv<64><<<NPT / 4, 128, 0, stream>>>(x1h, idxb, c2w1h, c2b1, c2w2h, c2b2, x2h);

  // ---- EdgeConv 3 ----
  k_sqnorm<<<NPT / 256, 256, 0, stream>>>(x2h, sqbuf, NPT);
  k_knn64<<<64 * 64, 128, knn64_lds, stream>>>(x2h, sqbuf, idxb);
  k_edgeconv<64><<<NPT / 4, 128, 0, stream>>>(x2h, idxb, c3w1h, c3b1, c3w2h, c3b2, x3h);

  // ---- Final MLP + log_softmax ----
  k_mlp<<<NPT / 64, 128, 0, stream>>>(x1h, x2h, x3h, mw1h, mb1, mw2h, mb2,
                                      mw3h, mb3, mw4, mb4, out);

  (void)in_sizes; (void)n_in; (void)out_size; (void)ws_size;
}
